// MultiHeadAttention_45354854645921
// MI455X (gfx1250) — compile-verified, matrix operand order checked by reading, not run on hardware
//
#include <hip/hip_runtime.h>
#include <hip/hip_bf16.h>
#include <stdint.h>

typedef unsigned short u16;
typedef __attribute__((ext_vector_type(16))) __bf16 v16bf;
typedef __attribute__((ext_vector_type(8)))  float  v8f;

#define D_MODEL 2048
#define TSEQ    2048
#define NHEAD   16
#define DK      128
#define BATCH   2
#define MTOT    (BATCH*TSEQ) /* 4096 */
#define QK_SCALE 0.08838834764831845f /* 1/sqrt(128) */

// ---------- helpers ----------

static __device__ __forceinline__ u16 f2bf(float f) {
  uint32_t x = __float_as_uint(f);
  x += 0x7FFFu + ((x >> 16) & 1u);   // round-to-nearest-even
  return (u16)(x >> 16);
}

union FragU { uint4 u[2]; v16bf v; };

// Loads a 16x32 bf16 A/B fragment slice for one lane.
// Per ISA 7.12.2: lane holds row (lane&15); k = (lane>>4)*8 + {0..7} in v0..3
// and +16 in v4..7 -> two contiguous 16B chunks at p and p+16 elements.
static __device__ __forceinline__ v16bf load_frag(const u16* p) {
  FragU fr;
  fr.u[0] = *reinterpret_cast<const uint4*>(p);
  fr.u[1] = *reinterpret_cast<const uint4*>(p + 16);
  return fr.v;
}

static __device__ __forceinline__ v8f vzero8() {
  v8f z;
#pragma unroll
  for (int i = 0; i < 8; ++i) z[i] = 0.0f;
  return z;
}

static __device__ __forceinline__ v8f wmma_bf16(v16bf a, v16bf b, v8f c) {
  return __builtin_amdgcn_wmma_f32_16x16x32_bf16(false, a, false, b,
                                                 (short)0, c, false, false);
}

// CDNA5 async global->LDS copy, 16B per lane (tracked by ASYNCcnt).
// LDS dest address = low 32 bits of the generic pointer (ISA §10.2 aperture map).
static __device__ __forceinline__ void async_copy16(const u16* g, u16* l) {
  uint32_t loff = (uint32_t)(uintptr_t)l;
  asm volatile("global_load_async_to_lds_b128 %0, %1, off"
               :: "v"(loff), "v"(g) : "memory");
}

static __device__ __forceinline__ void wait_async0() {
  asm volatile("s_wait_asynccnt 0x0" ::: "memory");
}

// ---------- fp32 -> bf16 elementwise convert (x) ----------

__global__ __launch_bounds__(256) void cvt_kernel(const float* __restrict__ in,
                                                  u16* __restrict__ out, int n4) {
  int i = blockIdx.x * 256 + threadIdx.x;
  if (i >= n4) return;
  float4 v = reinterpret_cast<const float4*>(in)[i];
  uint2 o;
  o.x = (uint32_t)f2bf(v.x) | ((uint32_t)f2bf(v.y) << 16);
  o.y = (uint32_t)f2bf(v.z) | ((uint32_t)f2bf(v.w) << 16);
  reinterpret_cast<uint2*>(out)[i] = o;
}

// ---------- fp32 [K][N] -> bf16 [N][K] transpose-convert (weights) ----------

__global__ __launch_bounds__(256) void tconv_kernel(const float* __restrict__ W,
                                                    u16* __restrict__ Wt) {
  __shared__ float tile[32][33];
  int n0 = blockIdx.x * 32;
  int k0 = blockIdx.y * 32;
  int tx = threadIdx.x;   // 0..31
  int ty = threadIdx.y;   // 0..7
#pragma unroll
  for (int i = 0; i < 32; i += 8)
    tile[ty + i][tx] = W[(size_t)(k0 + ty + i) * D_MODEL + n0 + tx];
  __syncthreads();
#pragma unroll
  for (int i = 0; i < 32; i += 8)
    Wt[(size_t)(n0 + ty + i) * D_MODEL + k0 + tx] = f2bf(tile[tx][ty + i]);
}

// ---------- bf16 GEMM: C[M=4096][N=2048] = A[M][K=2048] * Bw[N][K]^T + bias ----------
// Double-buffered LDS; tiles staged with global_load_async_to_lds_b128.
// MODE 0: Q epilogue  -> bf16 [B,H,T,Dk], scaled by 1/sqrt(Dk)
// MODE 1: K epilogue  -> bf16 [B,H,T,Dk]
// MODE 2: V epilogue  -> bf16 [B,H,Dk,T]  (transposed, packed b128 stores)
// MODE 3: O epilogue  -> fp32 [M][N]      (final output)

template <int MODE>
__global__ __launch_bounds__(256) void gemm_kernel(const u16* __restrict__ A,
                                                   const u16* __restrict__ Bw,
                                                   const float* __restrict__ bias,
                                                   void* __restrict__ out) {
  // LDS tiles, rows padded to 40 bf16 (80B = 5*16B -> b128-aligned, bank spread)
  __shared__ __align__(16) u16 As[2][128 * 40];
  __shared__ __align__(16) u16 Bs[2][128 * 40];

  const int tid  = threadIdx.x;
  const int wave = tid >> 5;
  const int lane = tid & 31;
  const int l15  = lane & 15;
  const int g    = lane >> 4;

  const int mBase = (blockIdx.x & 31) * 128;   // 32 M tiles
  const int nBase = (blockIdx.x >> 5) * 128;   // 16 N tiles
  const int wm = wave & 3;                     // 4 waves along M
  const int wn = wave >> 2;                    // 2 waves along N

  // per-thread staging slot: 16B of A tile + 16B of B tile, twice
  const int row0 = tid >> 2;            // 0..63  (round 0), +64 (round 1)
  const int c8   = (tid & 3) * 8;       // 0,8,16,24

  auto stage = [&](int kt, int buf) {
#pragma unroll
    for (int r = 0; r < 2; ++r) {
      int row = row0 + r * 64;
      async_copy16(&A[(size_t)(mBase + row) * D_MODEL + kt * 32 + c8],
                   &As[buf][row * 40 + c8]);
      async_copy16(&Bw[(size_t)(nBase + row) * D_MODEL + kt * 32 + c8],
                   &Bs[buf][row * 40 + c8]);
    }
  };

  v8f acc[2][4];
#pragma unroll
  for (int mt = 0; mt < 2; ++mt)
#pragma unroll
    for (int nf = 0; nf < 4; ++nf) acc[mt][nf] = vzero8();

  stage(0, 0);
  wait_async0();
  __syncthreads();

  for (int kt = 0; kt < D_MODEL / 32; ++kt) {
    const int cur = kt & 1;
    if (kt + 1 < D_MODEL / 32) stage(kt + 1, cur ^ 1);  // prefetch next tile

    v16bf af[2], bf[4];
#pragma unroll
    for (int mt = 0; mt < 2; ++mt)
      af[mt] = load_frag(&As[cur][(wm * 32 + mt * 16 + l15) * 40 + g * 8]);
#pragma unroll
    for (int nf = 0; nf < 4; ++nf)
      bf[nf] = load_frag(&Bs[cur][(wn * 64 + nf * 16 + l15) * 40 + g * 8]);
#pragma unroll
    for (int mt = 0; mt < 2; ++mt)
#pragma unroll
      for (int nf = 0; nf < 4; ++nf)
        acc[mt][nf] = wmma_bf16(af[mt], bf[nf], acc[mt][nf]);

    wait_async0();     // next tile landed in the other buffer
    __syncthreads();   // everyone done reading cur / writing nxt
  }

  // epilogue: lane holds col n = nf*16 + l15, rows m = m0 + r (r=0..7)
#pragma unroll
  for (int mt = 0; mt < 2; ++mt) {
#pragma unroll
    for (int nf = 0; nf < 4; ++nf) {
      const int n  = nBase + wn * 64 + nf * 16 + l15;
      const float bv = bias[n];
      const int m0 = mBase + wm * 32 + mt * 16 + g * 8;
      float vals[8];
#pragma unroll
      for (int r = 0; r < 8; ++r) {
        float v = acc[mt][nf][r] + bv;
        if (MODE == 0) v *= QK_SCALE;
        vals[r] = v;
      }
      if (MODE == 0 || MODE == 1) {
        const int h = n >> 7, d = n & 127;
#pragma unroll
        for (int r = 0; r < 8; ++r) {
          int m = m0 + r;
          int b = m >> 11, t = m & 2047;
          ((u16*)out)[((size_t)(b * NHEAD + h) * TSEQ + t) * DK + d] = f2bf(vals[r]);
        }
      } else if (MODE == 2) {
        const int h = n >> 7, d = n & 127;
        const int b = m0 >> 11, t0 = m0 & 2047;  // 8 consecutive t, same b
        uint4 o;
        o.x = (uint32_t)f2bf(vals[0]) | ((uint32_t)f2bf(vals[1]) << 16);
        o.y = (uint32_t)f2bf(vals[2]) | ((uint32_t)f2bf(vals[3]) << 16);
        o.z = (uint32_t)f2bf(vals[4]) | ((uint32_t)f2bf(vals[5]) << 16);
        o.w = (uint32_t)f2bf(vals[6]) | ((uint32_t)f2bf(vals[7]) << 16);
        *reinterpret_cast<uint4*>(
            &((u16*)out)[((size_t)(b * NHEAD + h) * DK + d) * TSEQ + t0]) = o;
      } else {
#pragma unroll
        for (int r = 0; r < 8; ++r)
          ((float*)out)[(size_t)(m0 + r) * D_MODEL + n] = vals[r];
      }
    }
  }
}

// ---------- flash attention: Q[B,H,T,Dk], K[B,H,T,Dk], Vt[B,H,Dk,T] -> O[B,T,C] bf16 ----------

__global__ __launch_bounds__(256) void attn_kernel(const u16* __restrict__ Q,
                                                   const u16* __restrict__ K,
                                                   const u16* __restrict__ Vt,
                                                   u16* __restrict__ O) {
  __shared__ __align__(16) u16 Ks[64 * 136];      // [key][d], padded
  __shared__ __align__(16) u16 Vs[128 * 72];      // [d][key], padded
  __shared__ __align__(16) u16 Ps[8 * 16 * 72];   // per-wave P staging [row][key]

  const int tid  = threadIdx.x;
  const int wave = tid >> 5;
  const int lane = tid & 31;
  const int l15  = lane & 15;
  const int g    = lane >> 4;

  const int bh  = blockIdx.x >> 4;               // b*16+h
  const int qt  = blockIdx.x & 15;               // q tile (128 rows)
  const int qr0 = qt * 128 + wave * 16;          // this wave's 16 q rows

  // Q fragments held in registers for whole kernel (already scaled by 1/sqrt(Dk))
  const u16* Qrow = Q + (size_t)bh * TSEQ * DK + (size_t)(qr0 + l15) * DK;
  v16bf qf[4];
#pragma unroll
  for (int kd = 0; kd < 4; ++kd) qf[kd] = load_frag(Qrow + kd * 32 + g * 8);

  float mo[8], ls[8];
  v8f oacc[8];
#pragma unroll
  for (int r = 0; r < 8; ++r) { mo[r] = -3.0e38f; ls[r] = 0.0f; }
#pragma unroll
  for (int f = 0; f < 8; ++f) oacc[f] = vzero8();

  const size_t kBase = (size_t)bh * TSEQ * DK;   // K  [t][d]
  const size_t vBase = (size_t)bh * DK * TSEQ;   // Vt [d][t]

  for (int kt = 0; kt < TSEQ / 64; ++kt) {
    __syncthreads();   // protect K/V/P LDS reuse from previous iteration
    // stage K tile [64 keys][128 d] and Vt tile [128 d][64 keys] via async DMA
#pragma unroll
    for (int i = 0; i < 4; ++i) {
      int c  = i * 256 + tid;                    // 1024 chunks of 16B
      int kr = c >> 4, kc = (c & 15) * 8;
      async_copy16(&K[kBase + (size_t)(kt * 64 + kr) * DK + kc], &Ks[kr * 136 + kc]);
      int vr = c >> 3, vc = (c & 7) * 8;
      async_copy16(&Vt[vBase + (size_t)vr * TSEQ + kt * 64 + vc], &Vs[vr * 72 + vc]);
    }
    wait_async0();
    __syncthreads();

    // S = Q * K^T  (16 q rows x 64 keys per wave)
    v8f sacc[4];
#pragma unroll
    for (int nf = 0; nf < 4; ++nf) sacc[nf] = vzero8();
#pragma unroll
    for (int kd = 0; kd < 4; ++kd) {
#pragma unroll
      for (int nf = 0; nf < 4; ++nf) {
        v16bf kf = load_frag(&Ks[(nf * 16 + l15) * 136 + kd * 32 + g * 8]);
        sacc[nf] = wmma_bf16(qf[kd], kf, sacc[nf]);
      }
    }

    // online softmax: rows r (m = g*8+r) are split across the two 16-lane halves
    float cr[8];
#pragma unroll
    for (int r = 0; r < 8; ++r) {
      float mx = fmaxf(fmaxf(sacc[0][r], sacc[1][r]), fmaxf(sacc[2][r], sacc[3][r]));
#pragma unroll
      for (int off = 1; off < 16; off <<= 1)
        mx = fmaxf(mx, __shfl_xor(mx, off, 32));
      float mn = fmaxf(mo[r], mx);
      cr[r] = __expf(mo[r] - mn);
      mo[r] = mn;
      float ps = 0.0f;
#pragma unroll
      for (int nf = 0; nf < 4; ++nf) {
        float p = __expf(sacc[nf][r] - mn);
        sacc[nf][r] = p;
        ps += p;
      }
#pragma unroll
      for (int off = 1; off < 16; off <<= 1) ps += __shfl_xor(ps, off, 32);
      ls[r] = ls[r] * cr[r] + ps;
    }
#pragma unroll
    for (int f = 0; f < 8; ++f)
#pragma unroll
      for (int r = 0; r < 8; ++r) oacc[f][r] *= cr[r];

    // transpose P from D-layout to A-layout through per-wave LDS staging
#pragma unroll
    for (int nf = 0; nf < 4; ++nf)
#pragma unroll
      for (int r = 0; r < 8; ++r)
        Ps[wave * 16 * 72 + (g * 8 + r) * 72 + nf * 16 + l15] = f2bf(sacc[nf][r]);
    __syncthreads();

    // O += P * V   (B operand = Vs[d][t] rows, layout-native)
#pragma unroll
    for (int kd2 = 0; kd2 < 2; ++kd2) {
      v16bf pf = load_frag(&Ps[wave * 16 * 72 + l15 * 72 + kd2 * 32 + g * 8]);
#pragma unroll
      for (int f = 0; f < 8; ++f) {
        v16bf vf = load_frag(&Vs[(f * 16 + l15) * 72 + kd2 * 32 + g * 8]);
        oacc[f] = wmma_bf16(pf, vf, oacc[f]);
      }
    }
  }

  // normalize and write O[B,T,C] bf16
  const int b = bh >> 4, h = bh & 15;
  float rinv[8];
#pragma unroll
  for (int r = 0; r < 8; ++r) rinv[r] = 1.0f / ls[r];
#pragma unroll
  for (int f = 0; f < 8; ++f) {
    const int c = h * DK + f * 16 + l15;
#pragma unroll
    for (int r = 0; r < 8; ++r) {
      const int t = qr0 + g * 8 + r;
      O[((size_t)b * TSEQ + t) * D_MODEL + c] = f2bf(oacc[f][r] * rinv[r]);
    }
  }
}

// ---------- host launcher ----------

extern "C" void kernel_launch(void* const* d_in, const int* in_sizes, int n_in,
                              void* d_out, int out_size, void* d_ws, size_t ws_size,
                              hipStream_t stream) {
  (void)in_sizes; (void)n_in; (void)out_size; (void)ws_size;
  const float* x  = (const float*)d_in[0];
  const float* Wq = (const float*)d_in[1];
  const float* bq = (const float*)d_in[2];
  const float* Wk = (const float*)d_in[3];
  const float* bk = (const float*)d_in[4];
  const float* Wv = (const float*)d_in[5];
  const float* bv = (const float*)d_in[6];
  const float* Wo = (const float*)d_in[7];
  const float* bo = (const float*)d_in[8];

  char* ws = (char*)d_ws;
  const size_t MB = 1024ull * 1024ull;
  u16* xb  = (u16*)(ws +  0 * MB);   // 16 MB  x bf16 [M][K]
  u16* WqT = (u16*)(ws + 16 * MB);   //  8 MB  Wq^T bf16 [N][K]
  u16* WkT = (u16*)(ws + 24 * MB);
  u16* WvT = (u16*)(ws + 32 * MB);
  u16* WoT = (u16*)(ws + 40 * MB);
  u16* Qb  = (u16*)(ws + 48 * MB);   // 16 MB  [B,H,T,Dk]
  u16* Kb  = (u16*)(ws + 64 * MB);   // 16 MB  [B,H,T,Dk]
  u16* Vtb = (u16*)(ws + 80 * MB);   // 16 MB  [B,H,Dk,T]
  u16* Ob  = (u16*)(ws + 96 * MB);   // 16 MB  [B,T,C]

  cvt_kernel<<<(MTOT * D_MODEL / 4) / 256, 256, 0, stream>>>(x, xb, MTOT * D_MODEL / 4);
  dim3 tb(32, 8), tg(64, 64);
  tconv_kernel<<<tg, tb, 0, stream>>>(Wq, WqT);
  tconv_kernel<<<tg, tb, 0, stream>>>(Wk, WkT);
  tconv_kernel<<<tg, tb, 0, stream>>>(Wv, WvT);
  tconv_kernel<<<tg, tb, 0, stream>>>(Wo, WoT);

  gemm_kernel<0><<<512, 256, 0, stream>>>(xb, WqT, bq, (void*)Qb);
  gemm_kernel<1><<<512, 256, 0, stream>>>(xb, WkT, bk, (void*)Kb);
  gemm_kernel<2><<<512, 256, 0, stream>>>(xb, WvT, bv, (void*)Vtb);

  attn_kernel<<<512, 256, 0, stream>>>(Qb, Kb, Vtb, Ob);

  gemm_kernel<3><<<512, 256, 0, stream>>>(Ob, WoT, bo, d_out);
}
